// CompressedLinear_47656957116956
// MI455X (gfx1250) — matrix operand order repaired and checked by reading, not compiled, not run on hardware
//
#include <hip/hip_runtime.h>

typedef __attribute__((ext_vector_type(16))) _Float16 v16h;
typedef __attribute__((ext_vector_type(8)))  _Float16 v8h;
typedef __attribute__((ext_vector_type(8)))  float    v8f;
typedef __attribute__((ext_vector_type(4)))  float    v4f;
typedef __attribute__((ext_vector_type(4)))  int      v4i;
typedef __attribute__((ext_vector_type(4)))  unsigned u32x4;
typedef __attribute__((ext_vector_type(8)))  int      i32x8;
typedef __attribute__((ext_vector_type(4)))  int      i32x4;

#define GEMM_M 8192   // B*S = 4*2048
#define GEMM_N 4096   // OUT
#define GEMM_K 4096   // IN

#define BM 256
#define BN 256
#define BK 64
#define LDA 72        // padded LDS row stride (halves): conflict-free b128
#define LDB 72        // matches TDM pad: 128B data + 16B pad per row
#define NTHREADS 512  // 16 waves (wave32), 4x4 wave grid, 64x64 per wave

// ---------------------------------------------------------------------------
// Pass 1: dequantize weight to f16 workspace: w16[o,i] = (f16)(wq[o,i]-zp[o]).
// (wq - zp) in [-255,255] is exact in f16; scale/bias applied in epilogue.
// ---------------------------------------------------------------------------
__global__ __launch_bounds__(256)
void dequant_w_kernel(const int* __restrict__ wq, const int* __restrict__ zp,
                      _Float16* __restrict__ w16)
{
    const size_t idx = (size_t)blockIdx.x * 256 + threadIdx.x;
    const int o = (int)(idx >> 9);          // 512 threads per row of 4096
    const int i = ((int)idx & 511) << 3;    // 8 elements per thread
    const int z = zp[o];
    const size_t off = (size_t)o * GEMM_K + i;
    v4i q0 = *(const v4i*)(wq + off);
    v4i q1 = *(const v4i*)(wq + off + 4);
    v8h h;
#pragma unroll
    for (int t = 0; t < 4; ++t) {
        h[t]     = (_Float16)(float)(q0[t] - z);
        h[t + 4] = (_Float16)(float)(q1[t] - z);
    }
    *(v8h*)(w16 + off) = h;
}

// ---------------------------------------------------------------------------
// Pass 2: 256x256 block, 16 waves, 64x64 wave tile, BK=64, f16 WMMA.
// A (x): global->reg->cvt f16->LDS.  B (w16): Tensor Data Mover DMA with
// hardware LDS padding (row = 128B data + 16B pad -> stride 72 halves).
// ---------------------------------------------------------------------------
__global__ __launch_bounds__(NTHREADS)
void wmma_gemm_kernel(const float* __restrict__ x, const _Float16* __restrict__ w16,
                      const float* __restrict__ scale, const float* __restrict__ bias,
                      float* __restrict__ out)
{
    __shared__ __align__(16) _Float16 sA[2][BM * LDA];   // 72 KB
    __shared__ __align__(16) _Float16 sB[2][BN * LDB];   // 72 KB

    const int tid      = threadIdx.x;
    const int lane     = tid & 31;
    const int wave     = tid >> 5;       // 0..15
    const int lane16   = lane & 15;
    const int laneHalf = lane >> 4;      // 0 or 1

    const int wave_m = wave >> 2;        // 0..3 : 64 rows of M
    const int wave_n = wave & 3;         // 0..3 : 64 cols of N

    const int blockM = blockIdx.y * BM;
    const int blockN = blockIdx.x * BN;

    // A staging map: 4 threads/row, 16 floats each, 128 rows/pass, 2 passes
    const int arow = tid >> 2;           // 0..127
    const int acol = (tid & 3) << 4;     // 0,16,32,48 (elements)
    const float* aptr0 = x + (size_t)(blockM + arow) * GEMM_K + acol;
    const float* aptr1 = aptr0 + (size_t)128 * GEMM_K;

    v4f aR[4];                           // 16 floats of in-flight A data
    v8f acc[4][4] = {};                  // 128 accumulator VGPRs

    // ---- TDM: one descriptor DMAs the whole 256x64 f16 B tile into LDS ----
    auto tdmB = [&](int kt, int buf) {
        const unsigned lds = (unsigned)(size_t)&sB[buf][0];
        const unsigned long long ga = (unsigned long long)(size_t)w16 +
            ((size_t)blockN * GEMM_K + (size_t)kt * BK) * sizeof(_Float16);
        u32x4 g0;
        g0[0] = 1u;                                   // count=1, user mode
        g0[1] = lds;                                  // lds_addr
        g0[2] = (unsigned)ga;                         // global_addr[31:0]
        g0[3] = (unsigned)((ga >> 32) & 0x1FFFFFFu)   // global_addr[56:32]
                | (2u << 30);                         // type=2 (image)
        i32x8 g1;
        g1[0] = (1 << 16)    // data_size = 2 bytes
              | (1 << 20)    // pad_enable
              | (4 << 22)    // pad_interval: 32 DWORDs (=128B row)
              | (3 << 25);   // pad_amount: 4 DWORDs (=16B pad)
        g1[1] = (GEMM_K & 0xFFFF) << 16;                          // tensor_dim0 lo
        g1[2] = ((GEMM_K >> 16) & 0xFFFF) | ((GEMM_N & 0xFFFF) << 16); // dim0 hi | dim1 lo
        g1[3] = ((GEMM_N >> 16) & 0xFFFF) | (BK << 16);           // dim1 hi | tile_dim0=64
        g1[4] = BN;                                               // tile_dim1=256, tile_dim2=0
        g1[5] = GEMM_K;                                           // tensor_dim0_stride lo32
        g1[6] = 0;
        g1[7] = 0;
        const i32x4 z4 = {0, 0, 0, 0};                // 2D tile: groups 2/3 null
        const i32x8 z8 = {0, 0, 0, 0, 0, 0, 0, 0};
        __builtin_amdgcn_tensor_load_to_lds(g0, g1, z4, z4, z8, 0);
    };

    auto loadA = [&](int kt, int pass) {
        const float* p = (pass ? aptr1 : aptr0) + (size_t)kt * BK;
#pragma unroll
        for (int i = 0; i < 4; ++i) aR[i] = *(const v4f*)(p + i * 4);
    };

    auto storeA = [&](int buf, int pass) {
        v8h h0, h1;
#pragma unroll
        for (int t = 0; t < 4; ++t) {
            h0[t]     = (_Float16)aR[0][t];
            h0[t + 4] = (_Float16)aR[1][t];
            h1[t]     = (_Float16)aR[2][t];
            h1[t + 4] = (_Float16)aR[3][t];
        }
        _Float16* d = &sA[buf][(arow + pass * 128) * LDA + acol];
        *(v8h*)d       = h0;
        *(v8h*)(d + 8) = h1;
    };

    auto compute_ks = [&](int buf, int ks) {   // one K=32 slice: 16 WMMAs
        v16h a[4], b[4];
        // A frag (ISA 7.12.2, 16-bit A 16x32): lanes 0-15 halves -> K 0..7,16..23
        // lanes 16-31 -> K 8..15,24..31
#pragma unroll
        for (int t = 0; t < 4; ++t) {
            const _Float16* p = &sA[buf][(wave_m * 64 + t * 16 + lane16) * LDA
                                         + ks * 32 + laneHalf * 8];
            v8h lo = *(const v8h*)p;
            v8h hi = *(const v8h*)(p + 16);
            a[t] = __builtin_shufflevector(lo, hi, 0,1,2,3,4,5,6,7,
                                                   8,9,10,11,12,13,14,15);
        }
        // B frag: N = lane%16; lanes 0-15 hold K 0..7,16..23, lanes 16-31 K 8..15,24..31
#pragma unroll
        for (int t = 0; t < 4; ++t) {
            const _Float16* p = &sB[buf][(wave_n * 64 + t * 16 + lane16) * LDB
                                         + ks * 32 + laneHalf * 8];
            v8h lo = *(const v8h*)p;
            v8h hi = *(const v8h*)(p + 16);
            b[t] = __builtin_shufflevector(lo, hi, 0,1,2,3,4,5,6,7,
                                                   8,9,10,11,12,13,14,15);
        }
#pragma unroll
        for (int i = 0; i < 4; ++i)
#pragma unroll
            for (int j = 0; j < 4; ++j)
                acc[i][j] = __builtin_amdgcn_wmma_f32_16x16x32_f16(
                    false, a[i], false, b[j], (short)0, acc[i][j], false, false);
    };

    // one pipeline stage: compute tile in bufC while staging tile ktNext->bufN
    auto stage = [&](int ktNext, int bufC, int bufN, bool doNext) {
        if (doNext) {
            if (wave == 0) tdmB(ktNext, bufN);     // DMA overlaps WMMAs below
            loadA(ktNext, 0);
        }
        compute_ks(bufC, 0);
        if (doNext) { storeA(bufN, 0); loadA(ktNext, 1); }
        compute_ks(bufC, 1);
        if (doNext) {
            storeA(bufN, 1);
            if (wave == 0) __builtin_amdgcn_s_wait_tensorcnt(0);
            __syncthreads();
        }
    };

    constexpr int T = GEMM_K / BK;       // 64 stages (even)

    // prologue: tile 0 into buffer 0
    if (wave == 0) tdmB(0, 0);
    loadA(0, 0); storeA(0, 0);
    loadA(0, 1); storeA(0, 1);
    if (wave == 0) __builtin_amdgcn_s_wait_tensorcnt(0);
    __syncthreads();

#pragma unroll 1
    for (int t = 0; t < T; t += 2) {
        stage(t + 1, 0, 1, true);        // T even: t+1 always valid
        stage(t + 2, 1, 0, t + 2 < T);
    }

    // Epilogue: C/D layout: VGPR r, lanes 0-15 -> M=r, N=lane;
    // lanes 16-31 -> M=8+r, N=lane-16. Apply per-channel scale + bias.
#pragma unroll
    for (int j = 0; j < 4; ++j) {
        const int n  = blockN + wave_n * 64 + j * 16 + lane16;
        const float sc = scale[n];
        const float bi = bias[n];
#pragma unroll
        for (int i = 0; i < 4; ++i) {
            const int m0 = blockM + wave_m * 64 + i * 16 + laneHalf * 8;
#pragma unroll
            for (int r = 0; r < 8; ++r) {
                const float v = acc[i][j][r] * sc + bi;
                __builtin_nontemporal_store(v, out + (size_t)(m0 + r) * GEMM_N + n);
            }
        }
    }
}

extern "C" void kernel_launch(void* const* d_in, const int* in_sizes, int n_in,
                              void* d_out, int out_size, void* d_ws, size_t ws_size,
                              hipStream_t stream) {
    (void)in_sizes; (void)n_in; (void)out_size; (void)ws_size;

    const float* x     = (const float*)d_in[0];   // [8192, 4096] f32
    const int*   wq    = (const int*)d_in[1];     // [4096, 4096] int32 (0..255)
    const int*   zp    = (const int*)d_in[2];     // [4096] int32
    const float* scale = (const float*)d_in[3];   // [4096] f32
    const float* bias  = (const float*)d_in[4];   // [4096] f32
    float*       out   = (float*)d_out;           // [8192, 4096] f32

    _Float16* w16 = (_Float16*)d_ws;              // 32 MB workspace

    const int dq_blocks = (GEMM_N * (GEMM_K / 8)) / 256;   // 8192
    dequant_w_kernel<<<dq_blocks, 256, 0, stream>>>(wq, zp, w16);

    dim3 grid(GEMM_N / BN, GEMM_M / BM);          // (16, 32)
    wmma_gemm_kernel<<<grid, NTHREADS, 0, stream>>>(x, w16, scale, bias, out);
}
